// CombinedLoss_83511344103780
// MI455X (gfx1250) — matrix operand order repaired and checked by reading, not compiled, not run on hardware
//
#include <hip/hip_runtime.h>
#include <hip/hip_bf16.h>

// ---------------- problem constants ----------------
#define BN   8192
#define CN   512
#define DN   256
#define ALPHA 0.5f
#define BETA  0.5f
#define GAMMA 0.5f
#define TAU   0.1f
#define EPS   1e-8f
// fixed logsumexp shift: |sim| <= (1/TAU)*(1+bf16 err) < 12, so exp(v-M0) never
// overflows and the 8191-term sum stays well inside f32 range.
#define M0   12.0f

// accumulator slots (floats at start of ws)
#define ACC_L1      0
#define ACC_L2      1
#define ACC_L3      2   // sum of mean_log_prob_pos over valid rows
#define ACC_VALID   3
#define ACC_L4      4
#define ACC_PRESENT 5

typedef __attribute__((ext_vector_type(16))) __bf16 v16bf;
typedef __attribute__((ext_vector_type(8)))  __bf16 v8bf;
typedef __attribute__((ext_vector_type(8)))  float  v8f;

union V16 { v16bf v; v8bf h[2]; };

// ---------------- reduction helpers (wave32) ----------------
__device__ inline float warpReduceSum(float v) {
  #pragma unroll
  for (int off = 16; off > 0; off >>= 1) v += __shfl_xor(v, off, 32);
  return v;
}
__device__ inline float warpReduceMax(float v) {
  #pragma unroll
  for (int off = 16; off > 0; off >>= 1) v = fmaxf(v, __shfl_xor(v, off, 32));
  return v;
}
__device__ inline float blockReduceSum(float v, float* sm) {
  int lane = threadIdx.x & 31, w = threadIdx.x >> 5;
  v = warpReduceSum(v);
  if (lane == 0) sm[w] = v;
  __syncthreads();
  float r = (threadIdx.x < (blockDim.x >> 5)) ? sm[threadIdx.x] : 0.0f;
  if (w == 0) { r = warpReduceSum(r); if (lane == 0) sm[0] = r; }
  __syncthreads();
  r = sm[0];
  __syncthreads();
  return r;
}
__device__ inline float blockReduceMax(float v, float* sm) {
  int lane = threadIdx.x & 31, w = threadIdx.x >> 5;
  v = warpReduceMax(v);
  if (lane == 0) sm[w] = v;
  __syncthreads();
  float r = (threadIdx.x < (blockDim.x >> 5)) ? sm[threadIdx.x] : -3.0e38f;
  if (w == 0) { r = warpReduceMax(r); if (lane == 0) sm[0] = r; }
  __syncthreads();
  r = sm[0];
  __syncthreads();
  return r;
}

// ---------------- kernels ----------------
__global__ void k_init(float* p, int n) {
  for (int i = blockIdx.x * blockDim.x + threadIdx.x; i < n;
       i += gridDim.x * blockDim.x) p[i] = 0.0f;
}

// per-class sums + counts (segment-sum) via atomics
__global__ void k_proto_accum(const float* __restrict__ emb,
                              const int* __restrict__ labels,
                              float* __restrict__ sums,
                              float* __restrict__ counts) {
  int i = blockIdx.x;                       // sample
  int lab = labels[i];
  int d = threadIdx.x;                      // 256 = DN
  atomicAdd(&sums[(size_t)lab * DN + d], emb[(size_t)i * DN + d]);
  if (d == 0) atomicAdd(&counts[lab], 1.0f);
}

// sums -> protos (in place), normalized protos -> bf16, present count
__global__ void k_finalize_protos(float* __restrict__ protos /* sums in */,
                                  const float* __restrict__ counts,
                                  __bf16* __restrict__ pnh,
                                  float* __restrict__ acc) {
  __shared__ float sm[8];
  int c = blockIdx.x, d = threadIdx.x;
  float cnt = counts[c];
  float p = protos[(size_t)c * DN + d] / fmaxf(cnt, 1.0f);
  protos[(size_t)c * DN + d] = p;
  float ss = blockReduceSum(p * p, sm);
  float inv = 1.0f / (sqrtf(ss) + EPS);
  pnh[(size_t)c * DN + d] = (__bf16)(p * inv);
  if (d == 0 && cnt > 0.0f) atomicAdd(&acc[ACC_PRESENT], 1.0f);
}

// z = emb / (||emb|| + eps) stored as bf16
__global__ void k_normalize_z(const float* __restrict__ emb,
                              __bf16* __restrict__ zh) {
  __shared__ float sm[8];
  int i = blockIdx.x, d = threadIdx.x;
  float e = emb[(size_t)i * DN + d];
  float ss = blockReduceSum(e * e, sm);
  float inv = 1.0f / (sqrtf(ss) + EPS);
  zh[(size_t)i * DN + d] = (__bf16)(e * inv);
}

// loss1: cross entropy, one block per row (512 logits, 256 threads)
__global__ void k_loss1(const float* __restrict__ logits,
                        const int* __restrict__ labels,
                        float* __restrict__ acc) {
  __shared__ float sm[8];
  int i = blockIdx.x, t = threadIdx.x;
  const float* row = logits + (size_t)i * CN;
  float v0 = row[t], v1 = row[t + 256];
  float m = blockReduceMax(fmaxf(v0, v1), sm);
  float s = blockReduceSum(__expf(v0 - m) + __expf(v1 - m), sm);
  if (t == 0) {
    float lse = m + __logf(s);
    atomicAdd(&acc[ACC_L1], lse - row[labels[i]]);
  }
}

// loss2: squared distance to own-class prototype
__global__ void k_loss2(const float* __restrict__ emb,
                        const float* __restrict__ protos,
                        const int* __restrict__ labels,
                        float* __restrict__ acc) {
  __shared__ float sm[8];
  int i = blockIdx.x, d = threadIdx.x;
  int lab = labels[i];
  float diff = emb[(size_t)i * DN + d] - protos[(size_t)lab * DN + d];
  float s = blockReduceSum(diff * diff, sm);
  if (d == 0) atomicAdd(&acc[ACC_L2], s);
}

// loss3: flash-style SupCon. One block = 16 rows; 8 waves stream 512 col tiles.
// sim tile 16x16 via 8x v_wmma_f32_16x16x32_bf16 (K=256); fixed-shift LSE.
__global__ __launch_bounds__(256) void k_supcon(const __bf16* __restrict__ zh,
                                                const int* __restrict__ labels,
                                                float* __restrict__ acc) {
  __shared__ __align__(16) __bf16 As[16 * DN];       // 8 KB: 16 rows of z
  __shared__ int labLDS[BN];                         // 32 KB: all labels
  __shared__ float tile[8][16][17];                  // per-wave sim tile
  __shared__ float redS[8][32], redP[8][32], redC[8][32];

  const int rowBase = blockIdx.x * 16;
  const int t = threadIdx.x;

  { // cooperative copy: 16-row A panel (bf16) + full label vector into LDS
    const uint32_t* src = (const uint32_t*)(zh + (size_t)rowBase * DN);
    uint32_t* dst = (uint32_t*)As;
    for (int k = t; k < 16 * DN / 2; k += 256) dst[k] = src[k];
    const uint4* ls = (const uint4*)labels;
    uint4* ld = (uint4*)labLDS;
    for (int k = t; k < BN / 4; k += 256) ld[k] = ls[k];
  }
  __syncthreads();

  const int wave = t >> 5;
  const int l = t & 31;
  const int r = l & 15;      // row (A) / column (B) within tile
  const int hi = l >> 4;     // half-wave select

  // A fragments for all 8 K-steps (ISA 16-bit A layout: lane r holds row r,
  // K = {hi*8 .. hi*8+7} and {16+hi*8 .. 16+hi*8+7} within each K32 step)
  V16 afrag[8];
  #pragma unroll
  for (int kk = 0; kk < 8; ++kk) {
    afrag[kk].h[0] = *(const v8bf*)(As + r * DN + kk * 32 + hi * 8);
    afrag[kk].h[1] = *(const v8bf*)(As + r * DN + kk * 32 + 16 + hi * 8);
  }

  const int myLab = labLDS[rowBase + r];
  const int gi = rowBase + r;
  const float invTau = 1.0f / TAU;

  float s = 0.0f, possum = 0.0f, pcnt = 0.0f;

  for (int ct = wave; ct < BN / 16; ct += 8) {
    const int colBase = ct * 16;
    v8f accf = {};
    // B layout: lane holds column n=r, K = {hi*8..hi*8+7} and {16+hi*8..16+hi*8+7}
    const __bf16* bbase = zh + (size_t)(colBase + r) * DN;
    #pragma unroll
    for (int kk = 0; kk < 8; ++kk) {
      V16 bfr;
      const v8bf* q = (const v8bf*)(bbase + kk * 32);
      bfr.h[0] = q[hi]; bfr.h[1] = q[2 + hi];
      accf = __builtin_amdgcn_wmma_f32_16x16x32_bf16(
          false, afrag[kk].v, false, bfr.v, (short)0, accf, false, false);
    }

    // prefetch next column tile's B panel (global_prefetch_b8) while we scan
    if (ct + 8 < BN / 16 && l < 16) {
      const char* nb = (const char*)(zh + (size_t)(colBase + 128 + r) * DN);
      __builtin_prefetch(nb, 0, 1);
      __builtin_prefetch(nb + 128, 0, 1);
      __builtin_prefetch(nb + 256, 0, 1);
      __builtin_prefetch(nb + 384, 0, 1);
    }

    // C/D layout: VGPR v, lane -> (M = v + hi*8, N = r)
    #pragma unroll
    for (int v = 0; v < 8; ++v) tile[wave][v + hi * 8][r] = accf[v] * invTau;
    asm volatile("s_wait_dscnt 0" ::: "memory");
    __builtin_amdgcn_wave_barrier();

    // all 32 lanes scan: lane (r, hi) owns row r, columns hi*8 .. hi*8+7
    #pragma unroll
    for (int c8 = 0; c8 < 8; ++c8) {
      int c = hi * 8 + c8;
      int j = colBase + c;
      float v = tile[wave][r][c];
      float e = __expf(v - M0);
      bool nd = (j != gi);                 // diagonal masked to NEG_INF
      if (nd) {
        s += e;
        if (labLDS[j] == myLab) { possum += v; pcnt += 1.0f; }
      }
    }
    __builtin_amdgcn_wave_barrier();
  }

  redS[wave][l] = s; redP[wave][l] = possum; redC[wave][l] = pcnt;
  __syncthreads();

  if (t < 16) {  // combine 8 waves x 2 half-wave partials per row
    float S = 0.0f, P = 0.0f, Cc = 0.0f;
    #pragma unroll
    for (int w = 0; w < 8; ++w) {
      S += redS[w][t] + redS[w][t + 16];
      P += redP[w][t] + redP[w][t + 16];
      Cc += redC[w][t] + redC[w][t + 16];
    }
    float lse = M0 + __logf(S);
    if (Cc > 0.0f) {
      float mlpp = (P - Cc * lse) / Cc;     // sum_pos(sim) - cnt*lse, averaged
      atomicAdd(&acc[ACC_L3], mlpp);
      atomicAdd(&acc[ACC_VALID], 1.0f);
    }
  }
}

// loss4: prototype Gram matrix via WMMA; sum of squared off-diagonal cosines
__global__ __launch_bounds__(256) void k_loss4(const __bf16* __restrict__ pnh,
                                               const float* __restrict__ counts,
                                               float* __restrict__ acc) {
  const int t = threadIdx.x;
  const int wave = t >> 5, l = t & 31, r = l & 15, hi = l >> 4;
  const int tileId = blockIdx.x * 8 + wave;        // 0..1023
  const int tileR = tileId >> 5, tileC = tileId & 31;

  const __bf16* arow = pnh + (size_t)(tileR * 16 + r) * DN;
  const __bf16* brow = pnh + (size_t)(tileC * 16 + r) * DN;
  v8f accf = {};
  #pragma unroll
  for (int kk = 0; kk < 8; ++kk) {
    V16 af, bfr;
    af.h[0] = *(const v8bf*)(arow + kk * 32 + hi * 8);
    af.h[1] = *(const v8bf*)(arow + kk * 32 + 16 + hi * 8);
    const v8bf* q = (const v8bf*)(brow + kk * 32);
    bfr.h[0] = q[hi]; bfr.h[1] = q[2 + hi];
    accf = __builtin_amdgcn_wmma_f32_16x16x32_bf16(
        false, af.v, false, bfr.v, (short)0, accf, false, false);
  }
  float local = 0.0f;
  const int gj = tileC * 16 + r;
  const bool pj = counts[gj] > 0.0f;
  #pragma unroll
  for (int v = 0; v < 8; ++v) {
    int gi = tileR * 16 + v + hi * 8;
    if (gi != gj && pj && counts[gi] > 0.0f) {
      float g = accf[v];
      local += g * g;
    }
  }
  local = warpReduceSum(local);
  if (l == 0 && local != 0.0f) atomicAdd(&acc[ACC_L4], local);
}

__global__ void k_finalize(const float* __restrict__ acc, float* __restrict__ out) {
  if (threadIdx.x == 0 && blockIdx.x == 0) {
    float loss1 = acc[ACC_L1] / (float)BN;
    float loss2 = acc[ACC_L2] / (float)BN;
    float loss3 = -acc[ACC_L3] / fmaxf(acc[ACC_VALID], 1.0f);
    float p = acc[ACC_PRESENT];
    float loss4 = acc[ACC_L4] / fmaxf(p * p - p, 1.0f);
    float total = loss1 + ALPHA * loss2 + BETA * loss3 + GAMMA * loss4;
    out[0] = total; out[1] = loss1; out[2] = loss2; out[3] = loss3; out[4] = loss4;
  }
}

// ---------------- launcher ----------------
extern "C" void kernel_launch(void* const* d_in, const int* in_sizes, int n_in,
                              void* d_out, int out_size, void* d_ws, size_t ws_size,
                              hipStream_t stream) {
  const float* logits = (const float*)d_in[0];   // [8192, 512]
  const float* emb    = (const float*)d_in[1];   // [8192, 256]
  const int*   labels = (const int*)d_in[2];     // [8192]
  float* out = (float*)d_out;                    // 5 floats

  char* ws = (char*)d_ws;
  float*  acc    = (float*)ws;                               // 8 floats (slot 0)
  float*  counts = (float*)(ws + 256);                       // 512 floats
  float*  protos = (float*)(ws + 4096);                      // 512*256 f32 (sums then protos)
  __bf16* pnh    = (__bf16*)(ws + 4096 + 524288);            // 512*256 bf16
  __bf16* zh     = (__bf16*)(ws + 4096 + 524288 + 262144);   // 8192*256 bf16
  // total ws usage: ~4.98 MB

  // zero acc + counts + proto sums (contiguous prefix of ws)
  const int zeroN = (4096 + 524288) / 4;
  k_init<<<(zeroN + 255) / 256, 256, 0, stream>>>((float*)ws, zeroN);

  k_proto_accum<<<BN, DN, 0, stream>>>(emb, labels, protos, counts);
  k_finalize_protos<<<CN, DN, 0, stream>>>(protos, counts, pnh, acc);
  k_normalize_z<<<BN, DN, 0, stream>>>(emb, zh);

  k_loss1<<<BN, 256, 0, stream>>>(logits, labels, acc);
  k_loss2<<<BN, DN, 0, stream>>>(emb, protos, labels, acc);
  k_supcon<<<BN / 16, 256, 0, stream>>>(zh, labels, acc);
  k_loss4<<<(CN / 16) * (CN / 16) / 8, 256, 0, stream>>>(pnh, counts, acc);

  k_finalize<<<1, 1, 0, stream>>>(acc, out);
}
